// SGNN_17927193494082
// MI455X (gfx1250) — matrix operand order repaired and checked by reading, not compiled, not run on hardware
//
#include <hip/hip_runtime.h>

#define BB 4
#define NN 4096
#define FF 64
#define NTILE 64                 // rows per workgroup / keys per j-tile
#define JT (NN / NTILE)
#define LDP 72                   // padded LDS row pitch (u16) to break bank aliasing

typedef unsigned short u16;
typedef __attribute__((ext_vector_type(16))) __bf16 v16bf;
typedef __attribute__((ext_vector_type(16))) unsigned short v16u;
typedef __attribute__((ext_vector_type(8))) float v8f;
typedef __attribute__((ext_vector_type(8))) unsigned int v8u;
typedef __attribute__((ext_vector_type(4))) unsigned int v4u;

union Frag { v16bf b; v16u u; v8u d; v4u q[2]; };

// Native f32 -> bf16.
__device__ __forceinline__ u16 f2bf(float f) {
  __bf16 b = (__bf16)f;
  return __builtin_bit_cast(u16, b);
}

#if __has_builtin(__builtin_amdgcn_cvt_pk_bf16_f32)
typedef __attribute__((ext_vector_type(2))) __bf16 v2bf;
__device__ __forceinline__ unsigned int pk2bf(float a, float b) {
  union { v2bf v; unsigned int u; } c;
  c.v = __builtin_amdgcn_cvt_pk_bf16_f32(a, b);
  return c.u;
}
#else
__device__ __forceinline__ unsigned int pk2bf(float a, float b) {
  return (unsigned int)f2bf(a) | ((unsigned int)f2bf(b) << 16);
}
#endif

// Raw hardware exp2 (v_exp_f32).
#if __has_builtin(__builtin_amdgcn_exp2f)
__device__ __forceinline__ float hexp2(float x) { return __builtin_amdgcn_exp2f(x); }
#else
__device__ __forceinline__ float hexp2(float x) { return exp2f(x); }
#endif

// Cross-half swap (lane ^ 16) of a packed dword: ds_swizzle SWAPX16 (xor=0x10,and=0x1f -> 0x401F).
__device__ __forceinline__ unsigned int swap16(unsigned int v) {
#if __has_builtin(__builtin_amdgcn_ds_swizzle)
  return (unsigned int)__builtin_amdgcn_ds_swizzle((int)v, 0x401F);
#else
  return (unsigned int)__shfl_xor((int)v, 16, 32);
#endif
}

// Async global -> LDS 16B copy (GLOBAL_LOAD_ASYNC_TO_LDS_B128, GV mode; ASYNCcnt-tracked).
// LDS operand is the low 32 bits of the generic pointer (LDS offset per ISA aperture rules).
__device__ __forceinline__ void async_copy16(const u16* g, u16* l) {
  asm volatile("global_load_async_to_lds_b128 %0, %1, off"
               :
               : "v"((unsigned int)(size_t)l), "v"(g)
               : "memory");
}
__device__ __forceinline__ void wait_async8() {
  asm volatile("s_wait_asynccnt 0x8" ::: "memory");
}
__device__ __forceinline__ void wait_async0() {
  asm volatile("s_wait_asynccnt 0x0" ::: "memory");
}

// A-fragment (16x32 bf16) from a row-major source: two contiguous 16B chunks per lane.
__device__ __forceinline__ v16bf afrag(const u16* base, int ld, int kbase, int lane) {
  const int m = lane & 15, h = lane >> 4;
  const u16* r = base + m * ld + kbase + h * 16;
  Frag f;
  f.q[0] = *(const v4u*)(r);
  f.q[1] = *(const v4u*)(r + 8);
  return f.b;
}

// Scalar-converting variant from f32 (used once per wave for x in kernel 1).
__device__ __forceinline__ v16bf afrag_f32(const float* base, int ld, int kbase, int lane) {
  const int m = lane & 15, h = lane >> 4;
  Frag f;
#pragma unroll
  for (int e = 0; e < 16; ++e) {
    const int k = kbase + h * 16 + e;
    f.u[e] = f2bf(base[m * ld + k]);
  }
  return f.b;
}

// B-fragment (32x16) from a K-contiguous source: B[k][n] = srcT[(col+n)*ldT + k].
__device__ __forceinline__ v16bf bfragKT(const u16* srcT, int ldT, int col, int kbase, int lane) {
  const int n = lane & 15, h = lane >> 4;
  const v4u* p = (const v4u*)(srcT + (col + n) * ldT + kbase + h * 16);
  Frag f;
  f.q[0] = p[0];
  f.q[1] = p[1];
  return f.b;
}

// Unpack->scale->repack a bf16 fragment (used once to fold log2e into q).
__device__ __forceinline__ v16bf scale_frag(v16bf in, float s) {
  Frag f, o;
  f.b = in;
#pragma unroll
  for (int e = 0; e < 16; ++e) {
    const float v = __uint_as_float((unsigned int)f.u[e] << 16) * s;
    o.u[e] = f2bf(v);
  }
  return o.b;
}

#define WMMA_BF16(A, B, C) \
  __builtin_amdgcn_wmma_f32_16x16x32_bf16(false, (A), false, (B), (short)0, (C), false, false)

// ============ Kernel 1: p = sigmoid(relu(x@w1+b1)@w2+b2) -> bf16 ; also x^T -> bf16 ============
__global__ void __launch_bounds__(128) k_mlp_p(
    const float* __restrict__ x, const float* __restrict__ w1, const float* __restrict__ b1,
    const float* __restrict__ w2, const float* __restrict__ b2,
    u16* __restrict__ p_bf, u16* __restrict__ x_bfT) {
  __shared__ __attribute__((aligned(16))) u16 sW1T[FF * LDP];
  __shared__ __attribute__((aligned(16))) u16 sW2T[FF * LDP];
  __shared__ __attribute__((aligned(16))) u16 sP[4 * 16 * LDP];

  const int t = threadIdx.x;
  const int lane = t & 31, wave = t >> 5;
  const int nlane = lane & 15, h = lane >> 4;
  const int rowbase = blockIdx.x * NTILE;  // flattened over B*N
  const int batch = rowbase >> 12;
  const int lrow = rowbase & (NN - 1);

  // Stage transposed bf16 weights (coalesced global reads; one-time scattered LDS writes).
  for (int i = t; i < FF * FF; i += 128) {
    const int k = i >> 6, n = i & 63;
    sW1T[n * LDP + k] = f2bf(w1[i]);
    sW2T[n * LDP + k] = f2bf(w2[i]);
  }

  // Produce bf16 x^T: x_bfT[batch][c][row] with packed 128-bit stores.
  {
    const int c = t >> 1, rh = (t & 1) * 32;
    u16* dst = x_bfT + (size_t)batch * FF * NN + (size_t)c * NN + lrow + rh;
#pragma unroll
    for (int kk = 0; kk < 32; kk += 8) {
      float v[8];
#pragma unroll
      for (int j = 0; j < 8; ++j) v[j] = x[(size_t)(rowbase + rh + kk + j) * FF + c];
      v4u pk;
      pk.x = pk2bf(v[0], v[1]);
      pk.y = pk2bf(v[2], v[3]);
      pk.z = pk2bf(v[4], v[5]);
      pk.w = pk2bf(v[6], v[7]);
      *(v4u*)(dst + kk) = pk;
    }
  }
  __syncthreads();

  const int qrow = rowbase + wave * 16;
  v16bf ax0 = afrag_f32(x + (size_t)qrow * FF, FF, 0, lane);
  v16bf ax1 = afrag_f32(x + (size_t)qrow * FF, FF, 32, lane);

  u16* tw = sP + wave * 16 * LDP;
  // Layer 1: h = relu(x@w1 + b1) -> per-wave LDS region (row-major)
#pragma unroll
  for (int nt = 0; nt < 4; ++nt) {
    v8f acc = {};
    acc = WMMA_BF16(ax0, bfragKT(sW1T, LDP, nt * 16, 0, lane), acc);
    acc = WMMA_BF16(ax1, bfragKT(sW1T, LDP, nt * 16, 32, lane), acc);
    const float bias = b1[nt * 16 + nlane];
#pragma unroll
    for (int r = 0; r < 8; ++r) {
      float v = acc[r] + bias;
      v = v > 0.f ? v : 0.f;
      tw[(r + 8 * h) * LDP + nt * 16 + nlane] = f2bf(v);
    }
  }
  v16bf ah0 = afrag(tw, LDP, 0, lane);
  v16bf ah1 = afrag(tw, LDP, 32, lane);
  // Layer 2 + sigmoid (overwrite per-wave region with p)
#pragma unroll
  for (int nt = 0; nt < 4; ++nt) {
    v8f acc = {};
    acc = WMMA_BF16(ah0, bfragKT(sW2T, LDP, nt * 16, 0, lane), acc);
    acc = WMMA_BF16(ah1, bfragKT(sW2T, LDP, nt * 16, 32, lane), acc);
    const float bias = b2[nt * 16 + nlane];
#pragma unroll
    for (int r = 0; r < 8; ++r) {
      const float z = acc[r] + bias;
      const float pv = __builtin_amdgcn_rcpf(1.0f + __expf(-z));  // sigmoid
      tw[(r + 8 * h) * LDP + nt * 16 + nlane] = f2bf(pv);
    }
  }
  __syncthreads();
  // Vectorized 64x64 p-tile copy to global (b128 both sides; de-pad rows).
  {
    v4u* dst = (v4u*)(p_bf + (size_t)rowbase * FF);
#pragma unroll
    for (int j = 0; j < 4; ++j) {
      const int i = t + 128 * j;
      dst[i] = *((const v4u*)(sP + (i >> 3) * LDP) + (i & 7));
    }
  }
}

// ===== Kernel 2: out = relu(((softmax(p p^T) @ x + x)/2) @ wf + bf) =====
// Flash-style; S computed *transposed* (keys in VGPRs, queries in lanes): softmax row-sum is
// per-lane scalar, P is rebuilt into the O^T WMMA B-operand purely in registers
// (v_cvt_pk_bf16 pairs + ds_swizzle SWAPX16 + half-uniform selects). Tiles are staged with
// GLOBAL_LOAD_ASYNC_TO_LDS_B128 into double-buffered LDS (ASYNCcnt pipeline).
__global__ void __launch_bounds__(128) k_attn(
    const float* __restrict__ x, const float* __restrict__ wf, const float* __restrict__ bfv,
    const u16* __restrict__ p_bf, const u16* __restrict__ x_bfT,
    float* __restrict__ out) {
  __shared__ __attribute__((aligned(16))) u16 sPJ[2][NTILE * LDP];   // p_j tiles (double buffer)
  __shared__ __attribute__((aligned(16))) u16 sXJT[2][FF * LDP];     // x_j^T tiles (double buffer)
  __shared__ __attribute__((aligned(16))) u16 sT[4 * 16 * LDP];      // per-wave epilogue scratch

  const int t = threadIdx.x;
  const int lane = t & 31, wave = t >> 5;
  const int nlane = lane & 15, h = lane >> 4;
  const bool hi = h != 0;
  const int batch = blockIdx.x >> 6;
  const int tile = blockIdx.x & 63;
  const int qrow = batch * NN + tile * NTILE + wave * 16;  // flattened row index

  const u16* pB = p_bf + (size_t)batch * NN * FF;
  const u16* xT = x_bfT + (size_t)batch * FF * NN;

  // Loop-invariant q^T B-fragments, pre-scaled by log2(e); exp shift folded into WMMA C.
  const float LOG2E = 1.4426950408889634f;
  const float CSHIFT = -32.0f * LOG2E;
  const u16* pQ = p_bf + (size_t)qrow * FF;
  v16bf bq0 = scale_frag(bfragKT(pQ, FF, 0, 0, lane), LOG2E);
  v16bf bq1 = scale_frag(bfragKT(pQ, FF, 0, 32, lane), LOG2E);
  const v8f cinit = {CSHIFT, CSHIFT, CSHIFT, CSHIFT, CSHIFT, CSHIFT, CSHIFT, CSHIFT};

  v8f oT[4];  // O^T tiles: cols in VGPR index, queries in lanes
  {
    v8f z = {};
#pragma unroll
    for (int i = 0; i < 4; ++i) oT[i] = z;
  }
  float Ll = 0.f;  // per-lane partial softmax denominator (keys are lane-local)

  // Issue async loads for a j-tile into LDS buffer `buf` (8 x b128 per thread).
  auto issue_tile = [&](int jb, int buf) {
#pragma unroll
    for (int j = 0; j < 4; ++j) {
      const int i = t + 128 * j;
      async_copy16(pB + (size_t)jb * FF + i * 8,
                   &sPJ[buf][(i >> 3) * LDP + (i & 7) * 8]);
      async_copy16(xT + (size_t)(i >> 3) * NN + jb + ((i & 7) << 3),
                   &sXJT[buf][(i >> 3) * LDP + (i & 7) * 8]);
    }
  };

  issue_tile(0, 0);  // prime the pipeline

  for (int jt = 0; jt < JT; ++jt) {
    __syncthreads();  // all waves done reading the buffer we are about to refill
    if (jt + 1 < JT) {
      issue_tile((jt + 1) * NTILE, (jt + 1) & 1);
      wait_async8();  // my 8 ops for tile jt complete (in-order); tile jt+1 still in flight
    } else {
      wait_async0();
    }
    __syncthreads();  // everyone's tile-jt async writes have landed

    const u16* cPJ = sPJ[jt & 1];
    const u16* cXJ = sXJT[jt & 1];

    // S^T = p_j @ q^T : 4 key-tiles x (16x16), K=64 -> 8 WMMAs; C preloaded with -32*log2e.
    v8f c[4];
#pragma unroll
    for (int nt = 0; nt < 4; ++nt) {
      v8f s = WMMA_BF16(afrag(cPJ + nt * 16 * LDP, LDP, 0, lane), bq0, cinit);
      s = WMMA_BF16(afrag(cPJ + nt * 16 * LDP, LDP, 32, lane), bq1, s);
      c[nt] = s;
    }

    // P = exp2(S') : bounded logits (p in (0,1) => safe); accumulate denominator per lane.
#pragma unroll
    for (int nt = 0; nt < 4; ++nt) {
      float rs = 0.f;
#pragma unroll
      for (int r = 0; r < 8; ++r) {
        const float pv = hexp2(c[nt][r]);
        c[nt][r] = pv;
        rs += pv;
      }
      Ll += rs;
    }

    // Pack P pairwise to bf16 dwords; swap halves; select into B-fragments.
    unsigned int pkv[4][4], swv[4][4];
#pragma unroll
    for (int nt = 0; nt < 4; ++nt)
#pragma unroll
      for (int d = 0; d < 4; ++d) {
        pkv[nt][d] = pk2bf(c[nt][2 * d], c[nt][2 * d + 1]);
        swv[nt][d] = swap16(pkv[nt][d]);
      }

#pragma unroll
    for (int g = 0; g < 2; ++g) {  // key groups of 32
      Frag Pb;
#pragma unroll
      for (int d = 0; d < 4; ++d) {
        Pb.d[d] = hi ? swv[2 * g + 1][d] : pkv[2 * g][d];
        Pb.d[d + 4] = hi ? pkv[2 * g + 1][d] : swv[2 * g][d];
      }
      // O^T += x_j^T @ P^T  (4 col-tiles share the P^T B-fragment)
#pragma unroll
      for (int nt = 0; nt < 4; ++nt)
        oT[nt] = WMMA_BF16(afrag(cXJ + nt * 16 * LDP, LDP, 32 * g, lane), Pb.b, oT[nt]);
    }
  }

  // Softmax denominator: combine the two key-halves (single cross-lane op).
  const float Lq = Ll + __shfl_xor(Ll, 16, 32);
  const float rcpL = __builtin_amdgcn_rcpf(Lq);

  // Stage wf^T bf16 (reuse sPJ[0]).
  __syncthreads();
  for (int i = t; i < FF * FF; i += 128) {
    const int k = i >> 6, n = i & 63;
    sPJ[0][n * LDP + k] = f2bf(wf[i]);
  }
  __syncthreads();

  // One-time O^T -> O transpose through LDS, fused with t = (O/L + x)/2.
  // (rowsum(softmax)+diag = 2 -> A = R/2, analytic.)
  u16* tw = sT + wave * 16 * LDP;
  const float* xq = x + (size_t)qrow * FF;
#pragma unroll
  for (int nt = 0; nt < 4; ++nt)
#pragma unroll
    for (int r = 0; r < 8; ++r) {
      const int col = nt * 16 + r + 8 * h;
      const float v = 0.5f * (oT[nt][r] * rcpL + xq[nlane * FF + col]);
      tw[nlane * LDP + col] = f2bf(v);
    }
  v16bf at0 = afrag(tw, LDP, 0, lane);
  v16bf at1 = afrag(tw, LDP, 32, lane);
#pragma unroll
  for (int nt = 0; nt < 4; ++nt) {
    v8f f = {};
    f = WMMA_BF16(at0, bfragKT(sPJ[0], LDP, nt * 16, 0, lane), f);
    f = WMMA_BF16(at1, bfragKT(sPJ[0], LDP, nt * 16, 32, lane), f);
    const float bias = bfv[nt * 16 + nlane];
#pragma unroll
    for (int r = 0; r < 8; ++r) {
      float v = f[r] + bias;
      v = v > 0.f ? v : 0.f;
      out[(size_t)(qrow + r + 8 * h) * FF + nt * 16 + nlane] = v;
    }
  }
}

extern "C" void kernel_launch(void* const* d_in, const int* in_sizes, int n_in,
                              void* d_out, int out_size, void* d_ws, size_t ws_size,
                              hipStream_t stream) {
  (void)in_sizes; (void)n_in; (void)out_size; (void)ws_size;
  const float* x  = (const float*)d_in[0];
  const float* w1 = (const float*)d_in[1];
  const float* b1 = (const float*)d_in[2];
  const float* w2 = (const float*)d_in[3];
  const float* b2 = (const float*)d_in[4];
  const float* wf = (const float*)d_in[5];
  const float* bf = (const float*)d_in[6];
  float* out = (float*)d_out;

  u16* p_bf  = (u16*)d_ws;                      // 2 MB bf16 p, row-major
  u16* x_bfT = p_bf + (size_t)BB * NN * FF;     // 2 MB bf16 x^T per batch

  const int blocks = (BB * NN) / NTILE;  // 256
  k_mlp_p<<<blocks, 128, 0, stream>>>(x, w1, b1, w2, b2, p_bf, x_bfT);
  k_attn<<<blocks, 128, 0, stream>>>(x, wf, bf, p_bf, x_bfT, out);
}
